// MACTitans_7653631721693
// MI455X (gfx1250) — compile-verified, matrix operand order checked by reading, not run on hardware
//
#include <hip/hip_runtime.h>
#include <hip/hip_bf16.h>

// Problem constants (match reference)
#define Bsz 2
#define Ssz 2048
#define Dsz 1024
#define Hh  16
#define DHh 64
#define Mtot (Bsz * Ssz)   // 4096 rows total

typedef __attribute__((ext_vector_type(16))) __bf16 v16bf;
typedef __attribute__((ext_vector_type(8)))  float  v8f;

// K index held by register r for 16-bit A (16x32) / B (n-major) fragments,
// per CDNA5 ISA 7.12.2: regs 0-3 -> K {0,2,4,6}(+8 hi), regs 4-7 -> K {16..22}(+8 hi),
// each 32-bit reg packs (k, k+1).
__device__ __forceinline__ int wk(int r, int hi) {
  return ((r < 4) ? (2 * r) : (16 + 2 * (r - 4))) + (hi ? 8 : 0);
}

// Load fragment where the K dimension is contiguous in memory with leading dim `ld`
// per (m or n) row. Works for A[m][k] row-major and for B[k][n] stored n-major.
__device__ __forceinline__ v16bf load_frag_contigK(const __bf16* p, int ld, int lane) {
  const int hi = (lane >> 4) & 1;
  const int mn = lane & 15;
  v16bf f;
#pragma unroll
  for (int r = 0; r < 8; ++r) {
    const int k = wk(r, hi);
    f[2 * r]     = p[mn * ld + k];
    f[2 * r + 1] = p[mn * ld + k + 1];
  }
  return f;
}

// Load B fragment from row-major B[k][n] (K strided by ld).
__device__ __forceinline__ v16bf load_frag_strideK(const __bf16* p, int ld, int lane) {
  const int hi = (lane >> 4) & 1;
  const int n  = lane & 15;
  v16bf f;
#pragma unroll
  for (int r = 0; r < 8; ++r) {
    const int k = wk(r, hi);
    f[2 * r]     = p[k * ld + n];
    f[2 * r + 1] = p[(k + 1) * ld + n];
  }
  return f;
}

__device__ __forceinline__ v8f wmma_bf16(v16bf a, v16bf b, v8f c) {
  return __builtin_amdgcn_wmma_f32_16x16x32_bf16(false, a, false, b, (short)0, c,
                                                 false, false);
}

__device__ __forceinline__ float rowmax16(float v) {
#pragma unroll
  for (int m = 1; m <= 8; m <<= 1) v = fmaxf(v, __shfl_xor(v, m, 32));
  return v;
}
__device__ __forceinline__ float rowsum16(float v) {
#pragma unroll
  for (int m = 1; m <= 8; m <<= 1) v += __shfl_xor(v, m, 32);
  return v;
}

// Async bf16 global->LDS copy of 16 bytes (8 elems), tracked by ASYNCcnt.
// LDS byte address = low 32 bits of the flat address (ISA: LDS aperture maps
// addr[31:0] directly to the LDS offset).
__device__ __forceinline__ void async_copy_b128(__bf16* lds_dst, const __bf16* gsrc) {
  const unsigned lds_addr = (unsigned)(uintptr_t)lds_dst;
  asm volatile("global_load_async_to_lds_b128 %0, %1, off"
               :: "v"(lds_addr), "v"(gsrc) : "memory");
}
__device__ __forceinline__ void wait_async0() {
  asm volatile("s_wait_asynccnt 0" ::: "memory");
}

// ---------------------------------------------------------------------------
// Kernel 1: Q/K/V projection + bias + RoPE.  grid=(N/128, M/128, 3), block=256.
// Waves in a 4x2 grid: each wave computes a 32x64 sub-tile (8 WMMAs / K-step).
// ---------------------------------------------------------------------------
__global__ __launch_bounds__(256)
void qkv_rope_kernel(const float* __restrict__ x,
                     const float* __restrict__ Wq, const float* __restrict__ bq,
                     const float* __restrict__ Wk, const float* __restrict__ bk,
                     const float* __restrict__ Wv, const float* __restrict__ bv,
                     const float* __restrict__ sintab,
                     __bf16* __restrict__ qout, __bf16* __restrict__ kout,
                     __bf16* __restrict__ vout) {
  const int which = blockIdx.z;                 // 0=q, 1=k, 2=v
  const float* W    = (which == 0) ? Wq : (which == 1) ? Wk : Wv;
  const float* bias = (which == 0) ? bq : (which == 1) ? bk : bv;
  __bf16*      dst  = (which == 0) ? qout : (which == 1) ? kout : vout;

  const int rowBase = blockIdx.y * 128;
  const int colBase = blockIdx.x * 128;

  __shared__ alignas(16) __bf16 As[128 * 32];   // A tile, row-major [m][k]
  __shared__ alignas(16) __bf16 Bs[128 * 32];   // B tile, n-major  [n][k]

  const int tid  = threadIdx.x;
  const int lane = tid & 31;
  const int w    = tid >> 5;
  const int wr   = w >> 1;                      // wave row group (32 rows)
  const int wc   = w & 1;                       // wave col group (64 cols)

  v8f c0[4] = {}, c1[4] = {};

  for (int k0 = 0; k0 < Dsz; k0 += 32) {
    // Stage A: 128x32 fp32 -> bf16, float4 (coalesced) reads.
#pragma unroll
    for (int j = 0; j < 4; ++j) {
      const int e   = (tid + 256 * j) * 4;
      const int row = e >> 5, kk = e & 31;
      const float4 xv =
          *reinterpret_cast<const float4*>(&x[(size_t)(rowBase + row) * Dsz + k0 + kk]);
      As[e + 0] = (__bf16)xv.x; As[e + 1] = (__bf16)xv.y;
      As[e + 2] = (__bf16)xv.z; As[e + 3] = (__bf16)xv.w;
    }
    // Stage B: 32x128 fp32 -> n-major bf16 [n][k]; float4 along n (coalesced),
    // 4-way scatter into LDS.
#pragma unroll
    for (int j = 0; j < 4; ++j) {
      const int q4 = tid + 256 * j;
      const int kk = q4 >> 5;
      const int n4 = (q4 & 31) * 4;
      const float4 wv =
          *reinterpret_cast<const float4*>(&W[(size_t)(k0 + kk) * Dsz + colBase + n4]);
      Bs[(n4 + 0) * 32 + kk] = (__bf16)wv.x;
      Bs[(n4 + 1) * 32 + kk] = (__bf16)wv.y;
      Bs[(n4 + 2) * 32 + kk] = (__bf16)wv.z;
      Bs[(n4 + 3) * 32 + kk] = (__bf16)wv.w;
    }
    __syncthreads();

    const v16bf a0 = load_frag_contigK(As + (wr * 32) * 32, 32, lane);
    const v16bf a1 = load_frag_contigK(As + (wr * 32 + 16) * 32, 32, lane);
#pragma unroll
    for (int t = 0; t < 4; ++t) {
      const v16bf bfr = load_frag_contigK(Bs + (wc * 64 + t * 16) * 32, 32, lane);
      c0[t] = wmma_bf16(a0, bfr, c0[t]);
      c1[t] = wmma_bf16(a1, bfr, c1[t]);
    }
    __syncthreads();
  }

  // Epilogue: bias (+ RoPE for q/k), write bf16.
  const int hi = lane >> 4, n = lane & 15;
#pragma unroll
  for (int g = 0; g < 2; ++g) {
    v8f* cc = g ? c1 : c0;
#pragma unroll
    for (int t = 0; t < 4; ++t) {
#pragma unroll
      for (int r = 0; r < 8; ++r) {
        const int m    = r + hi * 8;
        const int grow = rowBase + wr * 32 + g * 16 + m;
        const int gcol = colBase + wc * 64 + t * 16 + n;
        float val = cc[t][r] + bias[gcol];
        if (which < 2) {
          const int s  = grow & (Ssz - 1);           // position within sequence
          const int dh = gcol & (DHh - 1);           // index within head
          const int i  = dh >> 1;
          const float sv = sintab[s * DHh + i];      // sin_pos[s][dh]
          const float cv = sintab[s * DHh + 32 + i]; // cos_pos[s][dh]
          const float partner = __shfl_xor(val, 1, 32); // value at dh^1
          val = (dh & 1) ? (val * cv + partner * sv)
                         : (val * cv - partner * sv);
        }
        dst[(size_t)grow * Dsz + gcol] = (__bf16)val;
      }
    }
  }
}

// ---------------------------------------------------------------------------
// Kernel 2: causal flash attention. grid=(S/128, B*H), block=256 (8 waves).
// Wave w owns query rows [qstart+16w, qstart+16w+16). Keys in blocks of 32,
// truncated at the causal bound. K/V blocks staged by async global->LDS DMA.
// ---------------------------------------------------------------------------
__global__ __launch_bounds__(256)
void attn_kernel(const __bf16* __restrict__ q, const __bf16* __restrict__ k,
                 const __bf16* __restrict__ v, __bf16* __restrict__ ctxout) {
  const int qstart = blockIdx.x * 128;
  const int bh = blockIdx.y;
  const int b = bh / Hh, h = bh % Hh;
  const int tid = threadIdx.x, lane = tid & 31, w = tid >> 5;
  const int hi = lane >> 4, ln = lane & 15;

  __shared__ alignas(16) __bf16 Ks[32 * 64];        // [key][d] row-major
  __shared__ alignas(16) __bf16 Vs[32 * 64];        // [key][d] row-major
  __shared__ alignas(16) __bf16 Ps[8 * 16 * 32];    // per-wave P scratch [m][key]

  // Persistent Q fragments (d 0..31 and 32..63) for this wave's 16 rows.
  const size_t qbase = ((size_t)(b * Ssz + qstart + w * 16)) * Dsz + h * DHh;
  v16bf aq0, aq1;
#pragma unroll
  for (int r = 0; r < 8; ++r) {
    const int kk = wk(r, hi);
    aq0[2 * r]     = q[qbase + (size_t)ln * Dsz + kk];
    aq0[2 * r + 1] = q[qbase + (size_t)ln * Dsz + kk + 1];
    aq1[2 * r]     = q[qbase + (size_t)ln * Dsz + 32 + kk];
    aq1[2 * r + 1] = q[qbase + (size_t)ln * Dsz + 32 + kk + 1];
  }

  float mrun[8], lrun[8];
  v8f ctx[4] = {};
#pragma unroll
  for (int r = 0; r < 8; ++r) { mrun[r] = -3e38f; lrun[r] = 0.f; }

  const int nblocks = (qstart + 128) / 32;   // causal bound for this query block
  for (int kb = 0; kb < nblocks; ++kb) {
    const int ks0 = kb * 32;
    // Stage 32x64 K and V blocks: one async b128 per thread per array.
    {
      const int e   = tid * 8;                 // 8 bf16 = 16 bytes per thread
      const int key = e >> 6, d = e & 63;
      const size_t g = ((size_t)(b * Ssz + ks0 + key)) * Dsz + h * DHh + d;
      async_copy_b128(&Ks[e], &k[g]);
      async_copy_b128(&Vs[e], &v[g]);
      wait_async0();
    }
    __syncthreads();

    // Scores: 16x32 tile per wave (two 16x16 C fragments), K-dim = DH = 64.
    v8f s0 = {}, s1 = {};
    {
      const v16bf b0 = load_frag_contigK(Ks, 64, lane);             // keys 0-15, d 0-31
      const v16bf b1 = load_frag_contigK(Ks + 32, 64, lane);        // keys 0-15, d 32-63
      s0 = wmma_bf16(aq0, b0, s0);
      s0 = wmma_bf16(aq1, b1, s0);
      const v16bf b2 = load_frag_contigK(Ks + 16 * 64, 64, lane);   // keys 16-31
      const v16bf b3 = load_frag_contigK(Ks + 16 * 64 + 32, 64, lane);
      s1 = wmma_bf16(aq0, b2, s1);
      s1 = wmma_bf16(aq1, b3, s1);
    }

    // Online softmax; row stats are per-register (uniform across 16-lane half).
    __bf16* Pw = Ps + w * 512;
#pragma unroll
    for (int r = 0; r < 8; ++r) {
      const int m  = r + hi * 8;
      const int qg = qstart + w * 16 + m;
      float x0 = s0[r] * 0.125f;               // 1/sqrt(64)
      float x1 = s1[r] * 0.125f;
      if (ks0 + ln > qg)      x0 = -1e30f;     // causal mask
      if (ks0 + 16 + ln > qg) x1 = -1e30f;
      const float rmax  = rowmax16(fmaxf(x0, x1));
      const float mnew  = fmaxf(mrun[r], rmax);
      const float scale = __expf(mrun[r] - mnew);
      const float p0 = __expf(x0 - mnew);
      const float p1 = __expf(x1 - mnew);
      lrun[r] = lrun[r] * scale + rowsum16(p0 + p1);
      mrun[r] = mnew;
#pragma unroll
      for (int t = 0; t < 4; ++t) ctx[t][r] *= scale;
      // C layout -> row-major P in LDS (wave-private; same-wave DS ops in order)
      Pw[m * 32 + ln]      = (__bf16)p0;
      Pw[m * 32 + 16 + ln] = (__bf16)p1;
    }

    // ctx += P (16x32) @ V (32x64)
    {
      const v16bf ap = load_frag_contigK(Pw, 32, lane);
#pragma unroll
      for (int t = 0; t < 4; ++t) {
        const v16bf bvf = load_frag_strideK(Vs + t * 16, 64, lane);
        ctx[t] = wmma_bf16(ap, bvf, ctx[t]);
      }
    }
    __syncthreads();
  }

  // Normalize and store ctx (bf16) in [B*S][D] layout, head h at cols h*64.
#pragma unroll
  for (int r = 0; r < 8; ++r) {
    const float inv = 1.0f / lrun[r];
    const int m  = r + hi * 8;
    const int qg = qstart + w * 16 + m;
    const size_t rowoff = ((size_t)(b * Ssz + qg)) * Dsz + h * DHh;
#pragma unroll
    for (int t = 0; t < 4; ++t) {
      ctxout[rowoff + t * 16 + ln] = (__bf16)(ctx[t][r] * inv);
    }
  }
}

// ---------------------------------------------------------------------------
// Kernel 3: output projection ctx @ Wo + bo -> fp32. grid=(N/128, M/128).
// Same 128x128 tile / 4x2 wave shell as kernel 1.
// ---------------------------------------------------------------------------
__global__ __launch_bounds__(256)
void out_proj_kernel(const __bf16* __restrict__ ctx, const float* __restrict__ Wo,
                     const float* __restrict__ bo, float* __restrict__ out) {
  const int rowBase = blockIdx.y * 128;
  const int colBase = blockIdx.x * 128;

  __shared__ alignas(16) __bf16 As[128 * 32];
  __shared__ alignas(16) __bf16 Bs[128 * 32];

  const int tid = threadIdx.x;
  const int lane = tid & 31;
  const int w = tid >> 5;
  const int wr = w >> 1, wc = w & 1;

  v8f c0[4] = {}, c1[4] = {};

  for (int k0 = 0; k0 < Dsz; k0 += 32) {
    // A is already bf16: copy 8 elems (16B) per thread per j via vector loads.
#pragma unroll
    for (int j = 0; j < 2; ++j) {
      const int e = (tid + 256 * j) * 8;
      const int row = e >> 5, kk = e & 31;
      *reinterpret_cast<uint4*>(&As[e]) = *reinterpret_cast<const uint4*>(
          &ctx[(size_t)(rowBase + row) * Dsz + k0 + kk]);
    }
#pragma unroll
    for (int j = 0; j < 4; ++j) {
      const int q4 = tid + 256 * j;
      const int kk = q4 >> 5;
      const int n4 = (q4 & 31) * 4;
      const float4 wv =
          *reinterpret_cast<const float4*>(&Wo[(size_t)(k0 + kk) * Dsz + colBase + n4]);
      Bs[(n4 + 0) * 32 + kk] = (__bf16)wv.x;
      Bs[(n4 + 1) * 32 + kk] = (__bf16)wv.y;
      Bs[(n4 + 2) * 32 + kk] = (__bf16)wv.z;
      Bs[(n4 + 3) * 32 + kk] = (__bf16)wv.w;
    }
    __syncthreads();

    const v16bf a0 = load_frag_contigK(As + (wr * 32) * 32, 32, lane);
    const v16bf a1 = load_frag_contigK(As + (wr * 32 + 16) * 32, 32, lane);
#pragma unroll
    for (int t = 0; t < 4; ++t) {
      const v16bf bfr = load_frag_contigK(Bs + (wc * 64 + t * 16) * 32, 32, lane);
      c0[t] = wmma_bf16(a0, bfr, c0[t]);
      c1[t] = wmma_bf16(a1, bfr, c1[t]);
    }
    __syncthreads();
  }

  const int hi = lane >> 4, n = lane & 15;
#pragma unroll
  for (int g = 0; g < 2; ++g) {
    v8f* cc = g ? c1 : c0;
#pragma unroll
    for (int t = 0; t < 4; ++t) {
#pragma unroll
      for (int r = 0; r < 8; ++r) {
        const int m = r + hi * 8;
        const int grow = rowBase + wr * 32 + g * 16 + m;
        const int gcol = colBase + wc * 64 + t * 16 + n;
        out[(size_t)grow * Dsz + gcol] = cc[t][r] + bo[gcol];
      }
    }
  }
}

// ---------------------------------------------------------------------------
extern "C" void kernel_launch(void* const* d_in, const int* in_sizes, int n_in,
                              void* d_out, int out_size, void* d_ws, size_t ws_size,
                              hipStream_t stream) {
  (void)in_sizes; (void)n_in; (void)out_size; (void)ws_size;
  const float* x      = (const float*)d_in[0];
  const float* Wq     = (const float*)d_in[1];
  const float* bq     = (const float*)d_in[2];
  const float* Wk     = (const float*)d_in[3];
  const float* bk     = (const float*)d_in[4];
  const float* Wv     = (const float*)d_in[5];
  const float* bv     = (const float*)d_in[6];
  const float* Wo     = (const float*)d_in[7];
  const float* bo     = (const float*)d_in[8];
  const float* sintab = (const float*)d_in[9];
  // d_in[10] = attention_mask: causal, computed analytically in-kernel.
  float* out = (float*)d_out;

  // Workspace layout: q, k, v, ctx — each Mtot*Dsz bf16 (8 MB), 32 MB total.
  __bf16* qb = (__bf16*)d_ws;
  __bf16* kb = qb + (size_t)Mtot * Dsz;
  __bf16* vb = kb + (size_t)Mtot * Dsz;
  __bf16* cb = vb + (size_t)Mtot * Dsz;

  qkv_rope_kernel<<<dim3(Dsz / 128, Mtot / 128, 3), 256, 0, stream>>>(
      x, Wq, bq, Wk, bk, Wv, bv, sintab, qb, kb, vb);
  attn_kernel<<<dim3(Ssz / 128, Bsz * Hh), 256, 0, stream>>>(qb, kb, vb, cb);
  out_proj_kernel<<<dim3(Dsz / 128, Mtot / 128), 256, 0, stream>>>(cb, Wo, bo, out);
}
